// XTAttention_50792283242805
// MI455X (gfx1250) — matrix operand order repaired and checked by reading, not compiled, not run on hardware
//
#include <hip/hip_runtime.h>

#define DEV __device__ __forceinline__

typedef __attribute__((ext_vector_type(16))) __bf16 v16bf;
typedef __attribute__((ext_vector_type(8)))  float  v8f;
typedef __attribute__((ext_vector_type(4)))  unsigned u32x4;
typedef __attribute__((ext_vector_type(8)))  int      i32x8;
typedef __attribute__((ext_vector_type(4)))  int      i32x4;
typedef unsigned int   __attribute__((may_alias)) u32a;
typedef unsigned short __attribute__((may_alias)) u16a;
typedef u32x4          __attribute__((may_alias)) u32x4a;

union Frag16 { v16bf v; unsigned u[8]; u32x4 q[2]; };

DEV unsigned short f2bf(float x) {
    unsigned u = __builtin_bit_cast(unsigned, x);
    return (unsigned short)((u + 0x7FFFu + ((u >> 16) & 1u)) >> 16);
}

DEV v8f vzero() { v8f z = {0.f,0.f,0.f,0.f,0.f,0.f,0.f,0.f}; return z; }

DEV v8f wmma_bf16(Frag16 a, Frag16 b, v8f c) {
    // D(f32 16x16) = A(bf16 16x32) * B(bf16 32x16) + C
    return __builtin_amdgcn_wmma_f32_16x16x32_bf16(
        false, a.v, false, b.v, (short)0, c, false, false);
}

// Low 32 bits of a flat pointer to a __shared__ object are the LDS byte
// offset (ISA 10.2: LDS aperture).
DEV unsigned lds_off(const void* p) { return (unsigned)(size_t)p; }

// Async global->LDS copy, 16B per lane (ASYNCcnt).
DEV void async_load_b128(unsigned lds, const void* gptr) {
    asm volatile("global_load_async_to_lds_b128 %0, %1, off"
                 :: "v"(lds), "v"(gptr) : "memory");
}
DEV void wait_async0() { asm volatile("s_wait_asynccnt 0" ::: "memory"); }

// Tensor Data Mover: 2-D bf16 tile (tile_w x tile_h elems, row stride in
// elems) from global into LDS at lds_base. LDS rows are padded: after every
// 32 dwords written, skip 4 dwords (pad_interval code 4 = 32, amount code 3
// = 4) -> matches a [rows][36]-dword LDS layout for 64-elem bf16 rows.
DEV void tdm_load_2d_bf16(unsigned lds_base, const void* gaddr,
                          unsigned tile_w, unsigned tile_h, unsigned stride) {
    size_t ga = (size_t)gaddr;
    u32x4 g0;
    g0.x = 1u;                                            // count=1, is_restore=0
    g0.y = lds_base;                                      // lds_addr
    g0.z = (unsigned)(ga & 0xFFFFFFFFu);                  // global_addr[31:0]
    g0.w = (unsigned)((ga >> 32) & 0x01FFFFFFu) | (2u << 30);  // addr[56:32], type=2
    i32x8 g1;
    g1.s0 = (int)((1u << 16)      // data_size = 1 -> 2 bytes
                | (1u << 20)      // pad_enable
                | (4u << 22)      // pad_interval: 32 dwords
                | (3u << 25));    // pad_amount: 4 dwords
    g1.s1 = (int)(tile_w << 16);                          // tensor_dim0[15:0] @ [31:16]
    g1.s2 = (int)((tile_w >> 16) | (tile_h << 16));       // dim0 hi, tensor_dim1 lo
    g1.s3 = (int)((tile_h >> 16) | (tile_w << 16));       // dim1 hi, tile_dim0
    g1.s4 = (int)(tile_h & 0xFFFFu);                      // tile_dim1 (tile_dim2=0)
    g1.s5 = (int)stride;                                  // tensor_dim0_stride lo32
    g1.s6 = 0;                                            // stride hi, dim1_stride lo
    g1.s7 = 0;
    i32x4 z4 = {0, 0, 0, 0};
    i32x8 z8 = {0, 0, 0, 0, 0, 0, 0, 0};
    // 6-arg toolchain variant: (g0, g1, g2, g3, g4, cpol)
    __builtin_amdgcn_tensor_load_to_lds(g0, g1, z4, z4, z8, 0);
}

constexpr int KDIM = 768;   // model dim (also GEMM N and K)
constexpr int NSEQ = 2048;  // sequence length (q and kv)

// ---------------------------------------------------------------------------
// kc[n] = sum_k latent[k] * Wkc[k][n]
// ---------------------------------------------------------------------------
__global__ __launch_bounds__(256) void kc_kernel(const float* __restrict__ latent,
                                                 const float* __restrict__ Wkc,
                                                 float* __restrict__ kc) {
    int n = blockIdx.x * 256 + threadIdx.x;
    if (n < KDIM) {
        float s = 0.f;
        for (int k = 0; k < KDIM; ++k) s += latent[k] * Wkc[(size_t)k * KDIM + n];
        kc[n] = s;
    }
}

// ---------------------------------------------------------------------------
// Multi-source GEMM: out[M x 768] = sum_p X_p[M x 768] @ W_p[768 x 768] (+colbias)
// 256 threads = 8 waves (2x4), macro tile 128x128, each wave 64x32 (4x2 frags).
// ---------------------------------------------------------------------------
template <bool XBF16, bool OUTF32>
__global__ __launch_bounds__(256) void gemm_kernel(
    const void* __restrict__ x0v, const void* __restrict__ x1v, const void* __restrict__ x2v,
    const float* __restrict__ w0, const float* __restrict__ w1, const float* __restrict__ w2,
    int npairs, const float* __restrict__ colbias, void* __restrict__ outv)
{
    __shared__ alignas(16) unsigned lx[128][20];  // X tile [m][k/2], 16B-aligned rows
    __shared__ alignas(16) unsigned lw[128][17];  // W^T tile [n][k/2] +pad

    const int tid  = threadIdx.x;
    const int lane = tid & 31, wave = tid >> 5;
    const int wm = wave >> 2, wn = wave & 3;      // 2 x 4 wave grid
    const int half = lane >> 4, lr = lane & 15;
    const int bm = blockIdx.y * 128, bn = blockIdx.x * 128;

    v8f acc[4][2];
    #pragma unroll
    for (int i = 0; i < 4; ++i) { acc[i][0] = vzero(); acc[i][1] = vzero(); }

    for (int pair = 0; pair < npairs; ++pair) {
        const void*  xv = pair == 0 ? x0v : (pair == 1 ? x1v : x2v);
        const float* W  = pair == 0 ? w0  : (pair == 1 ? w1  : w2);

        for (int k0 = 0; k0 < KDIM; k0 += 32) {
            // ---- stage X tile (128 x 32) as bf16 pairs ----
            if (XBF16) {
                // bf16 source: pure copy -> async DMA to LDS, 16B chunks
                const u16a* X = (const u16a*)xv;
                for (int i = tid; i < 128 * 4; i += 256) {
                    int m = i >> 2, c = i & 3;
                    async_load_b128(lds_off(&lx[m][c * 4]),
                                    X + (size_t)(bm + m) * KDIM + k0 + c * 8);
                }
            } else {
                const float* X = (const float*)xv;
                for (int i = tid; i < 128 * 16; i += 256) {
                    int m = i >> 4, c = i & 15;
                    const float* p = X + (size_t)(bm + m) * KDIM + k0 + c * 2;
                    if (c == 0 && k0 + 32 < KDIM) __builtin_prefetch(p + 32, 0, 0);
                    lx[m][c] = (unsigned)f2bf(p[0]) | ((unsigned)f2bf(p[1]) << 16);
                }
            }
            // ---- stage W tile transposed: lw[n][k] = W[k0+k][bn+n] ----
            for (int i = tid; i < 128 * 16; i += 256) {
                int n = i & 127, c = i >> 7;
                float b0l = W[(size_t)(k0 + c * 2)     * KDIM + bn + n];
                float b1l = W[(size_t)(k0 + c * 2 + 1) * KDIM + bn + n];
                lw[n][c] = (unsigned)f2bf(b0l) | ((unsigned)f2bf(b1l) << 16);
            }
            if (XBF16) wait_async0();
            __syncthreads();

            // ---- B fragments: lane n = lr, dword v -> k = half*16 + 2v ----
            Frag16 bf[2];
            #pragma unroll
            for (int nt = 0; nt < 2; ++nt) {
                int n = wn * 32 + nt * 16 + lr;
                #pragma unroll
                for (int v = 0; v < 8; ++v) bf[nt].u[v] = lw[n][half * 8 + v];
            }
            // ---- A fragments + WMMA ----
            #pragma unroll
            for (int mt = 0; mt < 4; ++mt) {
                Frag16 af;
                int m = wm * 64 + mt * 16 + lr;
                #pragma unroll
                for (int v = 0; v < 8; ++v) {
                    int kk = half * 16 + v * 2;
                    af.u[v] = lx[m][kk >> 1];
                }
                acc[mt][0] = wmma_bf16(af, bf[0], acc[mt][0]);
                acc[mt][1] = wmma_bf16(af, bf[1], acc[mt][1]);
            }
            __syncthreads();
        }
    }

    // ---- epilogue: C layout row = r + half*8, col = lr ----
    #pragma unroll
    for (int mt = 0; mt < 4; ++mt)
        #pragma unroll
        for (int nt = 0; nt < 2; ++nt) {
            int gn = bn + wn * 32 + nt * 16 + lr;
            float bias = colbias ? colbias[gn] : 0.f;
            #pragma unroll
            for (int r = 0; r < 8; ++r) {
                int gm = bm + wm * 64 + mt * 16 + half * 8 + r;
                float val = acc[mt][nt][r] + bias;
                if (OUTF32) ((float*)outv)[(size_t)gm * KDIM + gn] = val;
                else        ((u16a*)outv)[(size_t)gm * KDIM + gn] = f2bf(val);
            }
        }
}

// ---------------------------------------------------------------------------
// Flash attention: grid (qblock=32, h=12, b=2), 128 threads = 4 waves.
// Each wave owns 16 query rows; kv chunk = 64. K chunk staged by the TDM.
// ---------------------------------------------------------------------------
__global__ __launch_bounds__(128) void attn_kernel(
    const unsigned short* __restrict__ Qb, const unsigned short* __restrict__ Kb,
    const unsigned short* __restrict__ Vb, unsigned short* __restrict__ Ob)
{
    constexpr float SCALE = 0.125f;  // 64^-0.5
    __shared__ alignas(16) unsigned lk[64][36];    // K chunk [kv][d/2], TDM-padded rows
    __shared__ alignas(16) unsigned lv[64][33];    // V^T     [d][kv/2]
    __shared__ alignas(16) unsigned lp[4][16][33]; // per-wave P [q][kv/2]

    const int tid = threadIdx.x, lane = tid & 31, wave = tid >> 5;
    const int half = lane >> 4, lr = lane & 15;
    const int qb = blockIdx.x, h = blockIdx.y, b = blockIdx.z;
    const int q0 = qb * 64 + wave * 16;
    const size_t rowQ = ((size_t)b * NSEQ + q0 + lr) * KDIM + h * 64;

    // Q fragments (A layout): per lane 16 contiguous k (k = 16*half + s) per chunk
    Frag16 qf[2];
    #pragma unroll
    for (int c = 0; c < 2; ++c) {
        qf[c].q[0] = *(const u32x4a*)(Qb + rowQ + c * 32 + half * 16);
        qf[c].q[1] = *(const u32x4a*)(Qb + rowQ + c * 32 + half * 16 + 8);
    }

    float m_i[8], l_i[8];
    v8f o[4];
    #pragma unroll
    for (int r = 0; r < 8; ++r) { m_i[r] = -1e30f; l_i[r] = 0.f; }
    #pragma unroll
    for (int t = 0; t < 4; ++t) o[t] = vzero();

    for (int kv0 = 0; kv0 < NSEQ; kv0 += 64) {
        // ---- K chunk via Tensor Data Mover (one wave issues the DMA) ----
        if (wave == 0) {
            tdm_load_2d_bf16(lds_off(&lk[0][0]),
                             Kb + ((size_t)b * NSEQ + kv0) * KDIM + h * 64,
                             /*tile_w=*/64, /*tile_h=*/64, /*stride=*/KDIM);
            __builtin_amdgcn_s_wait_tensorcnt(0);
        }
        // ---- stage V transposed: lv[d][kv] ----
        for (int i = tid; i < 64 * 64; i += 128) {
            int kv = i >> 6, d = i & 63;
            ((u16a*)&lv[d][0])[kv] = Vb[((size_t)b * NSEQ + kv0 + kv) * KDIM + h * 64 + d];
        }
        __syncthreads();

        // ---- S = Q @ K^T (4 kv sub-tiles, 2 k-chunks each) ----
        v8f s[4];
        #pragma unroll
        for (int j = 0; j < 4; ++j) {
            Frag16 b0, b1;
            #pragma unroll
            for (int v = 0; v < 8; ++v) {
                b0.u[v] = lk[j * 16 + lr][half * 8 + v];        // d 0..31
                b1.u[v] = lk[j * 16 + lr][16 + half * 8 + v];   // d 32..63
            }
            v8f z = vzero();
            z = wmma_bf16(qf[0], b0, z);
            z = wmma_bf16(qf[1], b1, z);
            s[j] = z;
        }

        // ---- online softmax (rows at r + half*8; 16 lanes per half) ----
        #pragma unroll
        for (int r = 0; r < 8; ++r) {
            float mx = -1e30f;
            #pragma unroll
            for (int j = 0; j < 4; ++j) mx = fmaxf(mx, s[j][r]);
            mx *= SCALE;
            mx = fmaxf(mx, __shfl_xor(mx, 1, 32));
            mx = fmaxf(mx, __shfl_xor(mx, 2, 32));
            mx = fmaxf(mx, __shfl_xor(mx, 4, 32));
            mx = fmaxf(mx, __shfl_xor(mx, 8, 32));
            float mnew  = fmaxf(m_i[r], mx);
            float alpha = __expf(m_i[r] - mnew);
            float rs = 0.f;
            #pragma unroll
            for (int j = 0; j < 4; ++j) {
                float p = __expf(s[j][r] * SCALE - mnew);
                rs += p;
                ((u16a*)&lp[wave][half * 8 + r][0])[j * 16 + lr] = f2bf(p);
            }
            rs += __shfl_xor(rs, 1, 32);
            rs += __shfl_xor(rs, 2, 32);
            rs += __shfl_xor(rs, 4, 32);
            rs += __shfl_xor(rs, 8, 32);
            l_i[r] = l_i[r] * alpha + rs;
            m_i[r] = mnew;
            #pragma unroll
            for (int t = 0; t < 4; ++t) o[t][r] *= alpha;
        }

        asm volatile("" ::: "memory");  // keep P stores before P loads (same-wave LDS in-order)

        // ---- O += P @ V ----
        Frag16 pf[2];
        #pragma unroll
        for (int c = 0; c < 2; ++c)
            #pragma unroll
            for (int v = 0; v < 8; ++v) {
                int kk = c * 32 + half * 16 + v * 2;
                pf[c].u[v] = lp[wave][lr][kk >> 1];
            }
        #pragma unroll
        for (int t = 0; t < 4; ++t) {
            Frag16 v0, v1;
            #pragma unroll
            for (int v = 0; v < 8; ++v) {
                v0.u[v] = lv[t * 16 + lr][half * 8 + v];        // kv 0..31
                v1.u[v] = lv[t * 16 + lr][16 + half * 8 + v];   // kv 32..63
            }
            o[t] = wmma_bf16(pf[0], v0, o[t]);
            o[t] = wmma_bf16(pf[1], v1, o[t]);
        }
        __syncthreads();
    }

    // ---- normalize and store (bf16, [b, nq, h*64+d] layout) ----
    #pragma unroll
    for (int r = 0; r < 8; ++r) {
        float inv = 1.f / l_i[r];
        size_t row = ((size_t)b * NSEQ + q0 + half * 8 + r) * KDIM + h * 64;
        #pragma unroll
        for (int t = 0; t < 4; ++t)
            Ob[row + t * 16 + lr] = f2bf(o[t][r] * inv);
    }
}

// ---------------------------------------------------------------------------
extern "C" void kernel_launch(void* const* d_in, const int* in_sizes, int n_in,
                              void* d_out, int out_size, void* d_ws, size_t ws_size,
                              hipStream_t stream)
{
    const float* words  = (const float*)d_in[0];
    const float* pwords = (const float*)d_in[1];
    const float* cwords = (const float*)d_in[2];
    const float* ckv    = (const float*)d_in[3];
    const float* pcross = (const float*)d_in[4];
    const float* Wq_w   = (const float*)d_in[5];
    const float* Wk_w   = (const float*)d_in[6];
    const float* Wv_w   = (const float*)d_in[7];
    const float* Wq_p   = (const float*)d_in[8];
    const float* Wk_p   = (const float*)d_in[9];
    const float* Wq_c   = (const float*)d_in[10];
    const float* Wk_c   = (const float*)d_in[11];
    const float* latent = (const float*)d_in[12];
    const float* Wo     = (const float*)d_in[13];
    const float* bo     = (const float*)d_in[14];

    const size_t M = 4096;                       // B * N rows
    const size_t matbytes = M * KDIM * sizeof(unsigned short);
    char* ws = (char*)d_ws;
    unsigned short* Qb = (unsigned short*)ws; ws += matbytes;
    unsigned short* Kb = (unsigned short*)ws; ws += matbytes;
    unsigned short* Vb = (unsigned short*)ws; ws += matbytes;
    unsigned short* Ob = (unsigned short*)ws; ws += matbytes;
    float* kc = (float*)ws;

    dim3 ggrid(KDIM / 128, M / 128);             // (6, 32)

    kc_kernel<<<3, 256, 0, stream>>>(latent, Wk_c, kc);

    // Q = words@Wq_w + pwords@Wq_p + cwords@Wq_c
    gemm_kernel<false, false><<<ggrid, 256, 0, stream>>>(
        words, pwords, cwords, Wq_w, Wq_p, Wq_c, 3, nullptr, Qb);
    // K = ckv@Wk_w + pcross@Wk_p + kc (broadcast over rows)
    gemm_kernel<false, false><<<ggrid, 256, 0, stream>>>(
        ckv, pcross, nullptr, Wk_w, Wk_p, nullptr, 2, kc, Kb);
    // V = ckv@Wv_w
    gemm_kernel<false, false><<<ggrid, 256, 0, stream>>>(
        ckv, nullptr, nullptr, Wv_w, nullptr, nullptr, 1, nullptr, Vb);

    attn_kernel<<<dim3(NSEQ / 64, 12, 2), 128, 0, stream>>>(Qb, Kb, Vb, Ob);

    // out = Ob @ Wo + bo  (f32 output)
    gemm_kernel<true, true><<<ggrid, 256, 0, stream>>>(
        Ob, nullptr, nullptr, Wo, nullptr, nullptr, 1, bo, d_out);
}
